// MHA_35167192219912
// MI455X (gfx1250) — matrix operand order repaired and checked by reading, not compiled, not run on hardware
//
#include <hip/hip_runtime.h>

#define N_EMBD  1024
#define N_HEAD  16
#define HEAD_DIM 64
#define B_SZ    4
#define T_SZ    2048
#define BT      (B_SZ * T_SZ)      // 8192 tokens
#define NEGV    (-1e10f)
#define LN_EPSF 1e-5f

typedef __attribute__((ext_vector_type(16))) __bf16 v16bf;
typedef __attribute__((ext_vector_type(8)))  float  v8f;
typedef __attribute__((ext_vector_type(4)))  int    v4i;

union FragBF { v16bf v; uint4 u[2]; };

#if __has_builtin(__builtin_amdgcn_global_load_async_to_lds_b128)
#define HAVE_ASYNC_LDS 1
#else
#define HAVE_ASYNC_LDS 0
#endif

__device__ __forceinline__ unsigned short f2bf(float f) {
  unsigned int u = __float_as_uint(f);
  u += 0x7fffu + ((u >> 16) & 1u);          // round-to-nearest-even
  return (unsigned short)(u >> 16);
}

// A fragment (16x32 bf16): lane row m = lane%16, half hf = lane/16.
// Elements: VGPR0-3 -> K = hf*8 + 0..7 ; VGPR4-7 -> K = 16 + hf*8 + 0..7
__device__ __forceinline__ v16bf frag_a(const unsigned short* row_k0, int hf) {
  FragBF f;
  const uint4* p = (const uint4*)row_k0;
  f.u[0] = p[hf];
  f.u[1] = p[2 + hf];
  return f.v;
}
// B fragment (32x16 bf16): lane col n = lane%16, half hf = lane/16.
// Elements: VGPR0-3 -> K = hf*8 + 0..7 ; VGPR4-7 -> K = 16 + hf*8 + 0..7
__device__ __forceinline__ v16bf frag_b(const unsigned short* row_k0, int hf) {
  FragBF f;
  const uint4* p = (const uint4*)row_k0;
  f.u[0] = p[hf];
  f.u[1] = p[2 + hf];
  return f.v;
}

#define WMMA_BF16(a, b, c) \
  __builtin_amdgcn_wmma_f32_16x16x32_bf16(false, (a), false, (b), (short)0, (c), false, false)

// Stage 16 bytes global -> LDS (async-tensor path when available)
__device__ __forceinline__ void stage16(const unsigned short* gsrc, unsigned short* ldst) {
#if HAVE_ASYNC_LDS
  __builtin_amdgcn_global_load_async_to_lds_b128(
      (__attribute__((address_space(1))) v4i*)(v4i*)gsrc,
      (__attribute__((address_space(3))) v4i*)(v4i*)ldst, 0, 0);
#else
  *(uint4*)ldst = *(const uint4*)gsrc;
#endif
}
__device__ __forceinline__ void stage_wait() {
#if HAVE_ASYNC_LDS
  asm volatile("s_wait_asynccnt 0x0" ::: "memory");
#endif
}

// ---------------------------------------------------------------------------
// Conversion kernels
// ---------------------------------------------------------------------------
__global__ void __launch_bounds__(256) cvt_bf16(const float* __restrict__ in,
                                                unsigned short* __restrict__ out, int n) {
  int i = blockIdx.x * 256 + threadIdx.x;
  if (i < n) out[i] = f2bf(in[i]);
}

// Wt[n*1024 + k] = bf16(W[k*1024 + n])   (pre-transpose for B operand)
__global__ void __launch_bounds__(256) transpose_bf16(const float* __restrict__ W,
                                                      unsigned short* __restrict__ Wt) {
  int i = blockIdx.x * 256 + threadIdx.x;
  int n = i >> 10, k = i & 1023;
  Wt[i] = f2bf(W[(size_t)k * N_EMBD + n]);
}

// ---------------------------------------------------------------------------
// Cooperative 64x64-tile GEMM body: 4 waves, B k-chunk staged in LDS (async),
// each wave owns a 16(M)x64(N) strip -> 4 WMMAs per k-step off one A fragment.
// Returns accumulators in acc[4]; n-column of lane = n0 + j*16 + lane16.
// ---------------------------------------------------------------------------
__device__ __forceinline__ void gemm64x64(
    const unsigned short* __restrict__ abase,   // A, row-major [*,1024]
    const unsigned short* __restrict__ wt,      // B^T, row-major [n][k]
    unsigned short* __restrict__ ldsB,          // [64][32] bf16, 4KB
    int m0, int n0, int lane16, int hf, int wslot, v8f acc[4]) {
  const unsigned short* arow = abase + (size_t)(m0 + wslot * 16 + lane16) * N_EMBD;
  const int tid = threadIdx.x;                 // 0..127
  for (int k0 = 0; k0 < N_EMBD; k0 += 32) {
    // stage B chunk [64 n x 32 k] (4KB): 256 x 16B, 2 per thread
#pragma unroll
    for (int p = 0; p < 2; ++p) {
      int idx = p * 128 + tid;                 // 0..255
      int nr = idx >> 2;                       // row within chunk (n)
      int c = idx & 3;                         // 16B sub-chunk along k
      stage16(wt + (size_t)(n0 + nr) * N_EMBD + k0 + c * 8, ldsB + nr * 32 + c * 8);
    }
    stage_wait();
    __syncthreads();
    v16bf a = frag_a(arow + k0, hf);
#pragma unroll
    for (int j = 0; j < 4; ++j) {
      v16bf b = frag_b(ldsB + (j * 16 + lane16) * 32, hf);
      acc[j] = WMMA_BF16(a, b, acc[j]);
    }
    __syncthreads();
  }
}

// ---------------------------------------------------------------------------
// Fused QKV projection: X[8192x1024] @ W -> Q/K [B,H,T,D], V transposed [B,H,D,T]
// One block per 64x64 output tile; which = 0:Q 1:K 2:V. n-tile of 64 == one head.
// ---------------------------------------------------------------------------
__global__ void __launch_bounds__(128) qkv_gemm(
    const unsigned short* __restrict__ xb,
    const unsigned short* __restrict__ wqt, const unsigned short* __restrict__ wkt,
    const unsigned short* __restrict__ wvt,
    const float* __restrict__ bq, const float* __restrict__ bk, const float* __restrict__ bv,
    unsigned short* __restrict__ Qb, unsigned short* __restrict__ Kb,
    unsigned short* __restrict__ Vtb) {
  __shared__ __align__(16) unsigned short ldsB[64 * 32];
  const int lane16 = threadIdx.x & 15;
  const int hf = (threadIdx.x >> 4) & 1;
  const int wslot = threadIdx.x >> 5;
  const int TM = BT / 64, TN = N_EMBD / 64;    // 128 x 16
  int gid = blockIdx.x;
  int which = gid / (TM * TN);
  int rem = gid % (TM * TN);
  int m0 = (rem / TN) * 64;
  int n0 = (rem % TN) * 64;

  const unsigned short* wt = (which == 0) ? wqt : (which == 1) ? wkt : wvt;
  const float* bias = (which == 0) ? bq : (which == 1) ? bk : bv;

  v8f acc[4];
#pragma unroll
  for (int j = 0; j < 4; ++j) acc[j] = (v8f){0.f, 0.f, 0.f, 0.f, 0.f, 0.f, 0.f, 0.f};
  gemm64x64(xb, wt, ldsB, m0, n0, lane16, hf, wslot, acc);

  int head = n0 >> 6;                          // whole 64-wide tile = one head
#pragma unroll
  for (int j = 0; j < 4; ++j) {
    int d = j * 16 + lane16;
    float bval = bias[n0 + d];
#pragma unroll
    for (int r = 0; r < 8; ++r) {
      int m = m0 + wslot * 16 + r + 8 * hf;
      int bb = m / T_SZ, t = m % T_SZ;
      unsigned short v = f2bf(acc[j][r] + bval);
      if (which == 2) {
        Vtb[(((size_t)bb * N_HEAD + head) * HEAD_DIM + d) * T_SZ + t] = v;
      } else if (which == 0) {
        Qb[(((size_t)bb * N_HEAD + head) * T_SZ + t) * HEAD_DIM + d] = v;
      } else {
        Kb[(((size_t)bb * N_HEAD + head) * T_SZ + t) * HEAD_DIM + d] = v;
      }
    }
  }
}

// ---------------------------------------------------------------------------
// Flash attention: one wave per (b, head, 16-row t-tile); stream s in 32-chunks.
// ---------------------------------------------------------------------------
__global__ void __launch_bounds__(128) attn_kernel(
    const unsigned short* __restrict__ Qb, const unsigned short* __restrict__ Kb,
    const unsigned short* __restrict__ Vtb, const int* __restrict__ mask,
    unsigned short* __restrict__ attb) {
  __shared__ __align__(16) unsigned short pbuf[4 * 16 * 32];  // per-wave P tile
  const int lane16 = threadIdx.x & 15;
  const int hf = (threadIdx.x >> 4) & 1;
  const int wslot = threadIdx.x >> 5;
  unsigned short* myp = pbuf + wslot * 512;

  const int TT = T_SZ / 16;                    // 128 t-tiles
  int gid = blockIdx.x * 4 + wslot;
  int tt = gid % TT;
  int head = (gid / TT) % N_HEAD;
  int b = gid / (TT * N_HEAD);
  int t0 = tt * 16;

  const unsigned short* Qbase = Qb + ((size_t)b * N_HEAD + head) * T_SZ * HEAD_DIM;
  const unsigned short* Kbase = Kb + ((size_t)b * N_HEAD + head) * T_SZ * HEAD_DIM;
  const unsigned short* Vtbase = Vtb + ((size_t)b * N_HEAD + head) * HEAD_DIM * T_SZ;

  // Q fragments held in registers for the whole s-loop (d chunks 0-31, 32-63)
  v16bf qa0 = frag_a(Qbase + (size_t)(t0 + lane16) * HEAD_DIM + 0, hf);
  v16bf qa1 = frag_a(Qbase + (size_t)(t0 + lane16) * HEAD_DIM + 32, hf);

  bool rm[8];
#pragma unroll
  for (int r = 0; r < 8; ++r)
    rm[r] = (mask[b * T_SZ + t0 + r + 8 * hf] == 0);

  float mrow[8], lrow[8];
#pragma unroll
  for (int r = 0; r < 8; ++r) { mrow[r] = -1e30f; lrow[r] = 0.f; }
  v8f o0 = {0.f, 0.f, 0.f, 0.f, 0.f, 0.f, 0.f, 0.f};
  v8f o1 = o0, o2 = o0, o3 = o0;
  const float sc = 0.125f;  // 1/sqrt(64)

  for (int s0 = 0; s0 < T_SZ; s0 += 32) {
    // S tile 16(t) x 32(s): two 16-col halves, K-dim = 64 (2 wmma each)
    v8f S0v = {0.f, 0.f, 0.f, 0.f, 0.f, 0.f, 0.f, 0.f};
    v8f S1v = S0v;
    {
      const unsigned short* kr0 = Kbase + (size_t)(s0 + lane16) * HEAD_DIM;
      S0v = WMMA_BF16(qa0, frag_b(kr0 + 0, hf), S0v);
      S0v = WMMA_BF16(qa1, frag_b(kr0 + 32, hf), S0v);
      const unsigned short* kr1 = Kbase + (size_t)(s0 + 16 + lane16) * HEAD_DIM;
      S1v = WMMA_BF16(qa0, frag_b(kr1 + 0, hf), S1v);
      S1v = WMMA_BF16(qa1, frag_b(kr1 + 32, hf), S1v);
    }
    // online softmax over this chunk; rows live per VGPR index, cols per lane
#pragma unroll
    for (int r = 0; r < 8; ++r) {
      float x0 = rm[r] ? NEGV : S0v[r] * sc;
      float x1 = rm[r] ? NEGV : S1v[r] * sc;
      float tmax = fmaxf(x0, x1);
      tmax = fmaxf(tmax, __shfl_xor(tmax, 1, 32));
      tmax = fmaxf(tmax, __shfl_xor(tmax, 2, 32));
      tmax = fmaxf(tmax, __shfl_xor(tmax, 4, 32));
      tmax = fmaxf(tmax, __shfl_xor(tmax, 8, 32));
      float nmax = fmaxf(mrow[r], tmax);
      float alpha = __expf(mrow[r] - nmax);
      float p0 = __expf(x0 - nmax);
      float p1 = __expf(x1 - nmax);
      float psum = p0 + p1;
      psum += __shfl_xor(psum, 1, 32);
      psum += __shfl_xor(psum, 2, 32);
      psum += __shfl_xor(psum, 4, 32);
      psum += __shfl_xor(psum, 8, 32);
      lrow[r] = lrow[r] * alpha + psum;
      mrow[r] = nmax;
      o0[r] *= alpha; o1[r] *= alpha; o2[r] *= alpha; o3[r] *= alpha;
      // stage P (bf16) to LDS row-major [16 x 32] for A-fragment reload
      int li = (r + 8 * hf) * 32 + lane16;
      myp[li] = f2bf(p0);
      myp[li + 16] = f2bf(p1);
    }
    v16bf pA = frag_a(myp + lane16 * 32, hf);   // DS -> A-fragment (layout swap)
    {
      const unsigned short* vr = Vtbase + (size_t)lane16 * T_SZ + s0;
      o0 = WMMA_BF16(pA, frag_b(vr, hf), o0);
      o1 = WMMA_BF16(pA, frag_b(vr + (size_t)16 * T_SZ, hf), o1);
      o2 = WMMA_BF16(pA, frag_b(vr + (size_t)32 * T_SZ, hf), o2);
      o3 = WMMA_BF16(pA, frag_b(vr + (size_t)48 * T_SZ, hf), o3);
    }
  }

#pragma unroll
  for (int r = 0; r < 8; ++r) {
    float inv = 1.0f / lrow[r];
    int t = t0 + r + 8 * hf;
    size_t rowoff = ((size_t)b * T_SZ + t) * N_EMBD + head * HEAD_DIM + lane16;
    attb[rowoff + 0]  = f2bf(o0[r] * inv);
    attb[rowoff + 16] = f2bf(o1[r] * inv);
    attb[rowoff + 32] = f2bf(o2[r] * inv);
    attb[rowoff + 48] = f2bf(o3[r] * inv);
  }
}

// ---------------------------------------------------------------------------
// Output projection + bias + residual: h = attb @ Wp + bp + x  (fp32 out)
// One block per 64x64 tile (cooperative, LDS-staged B).
// ---------------------------------------------------------------------------
__global__ void __launch_bounds__(128) proj_gemm(
    const unsigned short* __restrict__ attb, const unsigned short* __restrict__ wpt,
    const float* __restrict__ bp, const float* __restrict__ x,
    float* __restrict__ hbuf) {
  __shared__ __align__(16) unsigned short ldsB[64 * 32];
  const int lane16 = threadIdx.x & 15;
  const int hf = (threadIdx.x >> 4) & 1;
  const int wslot = threadIdx.x >> 5;
  const int TN = N_EMBD / 64;                  // 16
  int m0 = (blockIdx.x / TN) * 64;
  int n0 = (blockIdx.x % TN) * 64;

  v8f acc[4];
#pragma unroll
  for (int j = 0; j < 4; ++j) acc[j] = (v8f){0.f, 0.f, 0.f, 0.f, 0.f, 0.f, 0.f, 0.f};
  gemm64x64(attb, wpt, ldsB, m0, n0, lane16, hf, wslot, acc);

#pragma unroll
  for (int j = 0; j < 4; ++j) {
    int n = n0 + j * 16 + lane16;
    float bval = bp[n];
#pragma unroll
    for (int r = 0; r < 8; ++r) {
      int m = m0 + wslot * 16 + r + 8 * hf;
      size_t idx = (size_t)m * N_EMBD + n;
      hbuf[idx] = acc[j][r] + bval + x[idx];
    }
  }
}

// ---------------------------------------------------------------------------
// LayerNorm over last dim (1024) — one block per token row
// ---------------------------------------------------------------------------
__global__ void __launch_bounds__(256) layernorm_kernel(
    const float* __restrict__ hbuf, const float* __restrict__ g,
    const float* __restrict__ be, float* __restrict__ out) {
  __shared__ float ws[8], ws2[8];
  int row = blockIdx.x;
  const float* hr = hbuf + (size_t)row * N_EMBD;
  float s = 0.f, s2 = 0.f;
  for (int i = threadIdx.x; i < N_EMBD; i += 256) {
    float v = hr[i];
    s += v; s2 += v * v;
  }
  for (int m = 16; m; m >>= 1) {
    s  += __shfl_xor(s, m, 32);
    s2 += __shfl_xor(s2, m, 32);
  }
  int wid = threadIdx.x >> 5;
  if ((threadIdx.x & 31) == 0) { ws[wid] = s; ws2[wid] = s2; }
  __syncthreads();
  if (threadIdx.x < 32) {
    s  = (threadIdx.x < 8) ? ws[threadIdx.x]  : 0.f;
    s2 = (threadIdx.x < 8) ? ws2[threadIdx.x] : 0.f;
    for (int m = 4; m; m >>= 1) {
      s  += __shfl_xor(s, m, 32);
      s2 += __shfl_xor(s2, m, 32);
    }
    if (threadIdx.x == 0) { ws[0] = s; ws2[0] = s2; }
  }
  __syncthreads();
  float mu = ws[0] * (1.0f / N_EMBD);
  float var = ws2[0] * (1.0f / N_EMBD) - mu * mu;
  float rinv = rsqrtf(var + LN_EPSF);
  for (int i = threadIdx.x; i < N_EMBD; i += 256) {
    out[(size_t)row * N_EMBD + i] = (hr[i] - mu) * rinv * g[i] + be[i];
  }
}

// ---------------------------------------------------------------------------
extern "C" void kernel_launch(void* const* d_in, const int* in_sizes, int n_in,
                              void* d_out, int out_size, void* d_ws, size_t ws_size,
                              hipStream_t stream) {
  const float* x   = (const float*)d_in[0];
  const float* Wq  = (const float*)d_in[1];
  const float* bq  = (const float*)d_in[2];
  const float* Wk  = (const float*)d_in[3];
  const float* bk  = (const float*)d_in[4];
  const float* Wv  = (const float*)d_in[5];
  const float* bv  = (const float*)d_in[6];
  const float* Wp  = (const float*)d_in[7];
  const float* bp  = (const float*)d_in[8];
  const float* lng = (const float*)d_in[9];
  const float* lnb = (const float*)d_in[10];
  const int*   msk = (const int*)d_in[11];
  float* out = (float*)d_out;

  unsigned char* ws = (unsigned char*)d_ws;
  size_t off = 0;
  auto alloc = [&](size_t bytes) -> void* {
    void* p = ws + off;
    off += (bytes + 255) & ~(size_t)255;
    return p;
  };
  unsigned short* xb   = (unsigned short*)alloc((size_t)BT * N_EMBD * 2);       // 16 MiB
  unsigned short* wqt  = (unsigned short*)alloc((size_t)N_EMBD * N_EMBD * 2);   // 2 MiB
  unsigned short* wkt  = (unsigned short*)alloc((size_t)N_EMBD * N_EMBD * 2);
  unsigned short* wvt  = (unsigned short*)alloc((size_t)N_EMBD * N_EMBD * 2);
  unsigned short* wpt  = (unsigned short*)alloc((size_t)N_EMBD * N_EMBD * 2);
  unsigned short* Qb   = (unsigned short*)alloc((size_t)BT * N_EMBD * 2);       // [B,H,T,D]
  unsigned short* Kb   = (unsigned short*)alloc((size_t)BT * N_EMBD * 2);       // [B,H,T,D]
  unsigned short* Vtb  = (unsigned short*)alloc((size_t)BT * N_EMBD * 2);       // [B,H,D,T]
  unsigned short* attb = (unsigned short*)alloc((size_t)BT * N_EMBD * 2);       // [B,T,C]
  float*          hbuf = (float*)alloc((size_t)BT * N_EMBD * 4);                // 32 MiB

  // 1) precision conversion + weight transpose
  {
    int n = BT * N_EMBD;
    cvt_bf16<<<(n + 255) / 256, 256, 0, stream>>>(x, xb, n);
    int wn = N_EMBD * N_EMBD;
    transpose_bf16<<<wn / 256, 256, 0, stream>>>(Wq, wqt);
    transpose_bf16<<<wn / 256, 256, 0, stream>>>(Wk, wkt);
    transpose_bf16<<<wn / 256, 256, 0, stream>>>(Wv, wvt);
    transpose_bf16<<<wn / 256, 256, 0, stream>>>(Wp, wpt);
  }
  // 2) QKV projections (WMMA, cooperative 64x64 tiles, async LDS staging)
  {
    int blocks = (BT / 64) * (N_EMBD / 64) * 3;  // 6144
    qkv_gemm<<<blocks, 128, 0, stream>>>(xb, wqt, wkt, wvt, bq, bk, bv, Qb, Kb, Vtb);
  }
  // 3) flash attention (WMMA + LDS layout swap)
  {
    int waves = B_SZ * N_HEAD * (T_SZ / 16);     // 8192
    attn_kernel<<<waves / 4, 128, 0, stream>>>(Qb, Kb, Vtb, msk, attb);
  }
  // 4) output projection + residual (WMMA, cooperative 64x64 tiles)
  {
    int blocks = (BT / 64) * (N_EMBD / 64);      // 2048
    proj_gemm<<<blocks, 128, 0, stream>>>(attb, wpt, bp, x, hbuf);
  }
  // 5) LayerNorm
  layernorm_kernel<<<BT, 256, 0, stream>>>(hbuf, lng, lnb, out);
}
